// MultiheadSelfAttention_71536975282349
// MI455X (gfx1250) — matrix operand order repaired and checked by reading, not compiled, not run on hardware
//
#include <hip/hip_runtime.h>

typedef __attribute__((ext_vector_type(16))) _Float16 v16h;
typedef __attribute__((ext_vector_type(8)))  _Float16 v8h;
typedef __attribute__((ext_vector_type(8)))  float    v8f;

#define NB  4
#define NS  2048
#define ND  768
#define NH  12
#define NHD 64
#define KCH (ND / 32)   // 24 K-chunks of 32 for the projections

// ---------------------------------------------------------------------------
// fp32 -> fp16 conversion
// ---------------------------------------------------------------------------
__global__ void f32_to_f16_kernel(const float* __restrict__ src,
                                  _Float16* __restrict__ dst, int n) {
    int i = blockIdx.x * blockDim.x + threadIdx.x;
    if (i < n) dst[i] = (_Float16)src[i];
}

// ---------------------------------------------------------------------------
// Fragment loaders: A and B share one contiguous k order (not the ISA table order).
// A (16x32, f16): lane (lm = lane&15) = row M; hf = lane>>4.
//   a[i]: k = i + hf*16                  -> two contiguous 8-half loads.
// B (32x16, f16): lane lm = col N.
//   b[i]: k = i + hf*16                  -> one contiguous 16-half load.
// C/D (16x16, f32): c[r]: M = r + hf*8, N = lm.
// ---------------------------------------------------------------------------
__device__ inline v16h load_a_frag(const _Float16* row, int hf) {
    const v8h lo = *reinterpret_cast<const v8h*>(row + hf * 16);
    const v8h hi = *reinterpret_cast<const v8h*>(row + hf * 16 + 8);
    v16h a;
#pragma unroll
    for (int i = 0; i < 8; ++i) { a[i] = lo[i]; a[i + 8] = hi[i]; }
    return a;
}

__device__ inline v8f wmma_f16(v16h a, v16h b, v8f c) {
    return __builtin_amdgcn_wmma_f32_16x16x32_f16(false, a, false, b,
                                                  (short)0, c, false, false);
}

// ---------------------------------------------------------------------------
// QKV projection: qkv[k][m][o] = sum_d xh[m][d] * W_QKV[k][o][d]
// M = B*S = 8192, N = 768, K = 768.
// Each wave computes a 16x64 strip (4 accumulators, A reused 4x per chunk).
// Fully-unrolled K loop so the backend can keep many loads in flight.
// A 64-wide N strip == one head: strip index nt4 is the head id.
// Q,K stored [b][h][s][hd]; V stored transposed [b][h][hd][s].
// ---------------------------------------------------------------------------
__global__ __launch_bounds__(256) void qkv_gemm_kernel(
    const _Float16* __restrict__ xh, const _Float16* __restrict__ wh,
    _Float16* __restrict__ Qh, _Float16* __restrict__ Kh,
    _Float16* __restrict__ Vt) {
    const int lane = threadIdx.x & 31;
    const int wv   = threadIdx.x >> 5;
    const int hf   = lane >> 4;
    const int lm   = lane & 15;

    int tile = blockIdx.x * 8 + wv;          // 3 * 512 * 12 strips
    const int nt4 = tile % 12; tile /= 12;   // head index
    const int mt  = tile % 512;
    const int kk  = tile / 512;              // 0=Q 1=K 2=V

    const _Float16* arow = xh + (size_t)(mt * 16 + lm) * ND;
    const _Float16* bbase =
        wh + (size_t)kk * ND * ND + (size_t)(nt4 * 64 + lm) * ND;

    v8f acc[4];
#pragma unroll
    for (int n = 0; n < 4; ++n) acc[n] = (v8f){};

#pragma unroll
    for (int c = 0; c < KCH; ++c) {
        const v16h a = load_a_frag(arow + c * 32, hf);
#pragma unroll
        for (int n = 0; n < 4; ++n) {
            const v16h b = *reinterpret_cast<const v16h*>(
                bbase + (size_t)(n * 16) * ND + c * 32 + hf * 16);
            acc[n] = wmma_f16(a, b, acc[n]);
        }
    }

#pragma unroll
    for (int n = 0; n < 4; ++n) {
        const int ho = n * 16 + lm;          // column within the head
#pragma unroll
        for (int r = 0; r < 8; ++r) {
            const int m    = mt * 16 + r + hf * 8;   // global row in [0, B*S)
            const int bidx = m >> 11;
            const int sidx = m & (NS - 1);
            const _Float16 v = (_Float16)acc[n][r];
            const size_t bh = (size_t)(bidx * NH + nt4);
            if (kk == 0)      Qh[(bh * NS + sidx) * NHD + ho] = v;
            else if (kk == 1) Kh[(bh * NS + sidx) * NHD + ho] = v;
            else              Vt[(bh * NHD + ho) * NS + sidx] = v;
        }
    }
}

// ---------------------------------------------------------------------------
// RoPE in place on Q or K ([b][h][s][64]); pair i uses dims (2i, 2i+1).
// inv_freq = theta^(-(2i)/hd) = exp(-(i/32) * ln(10000))
// ---------------------------------------------------------------------------
__global__ void rope_kernel(_Float16* __restrict__ t,
                            const long long* __restrict__ pos) {
    const int idx = blockIdx.x * blockDim.x + threadIdx.x;  // B*H*S*32
    if (idx >= NB * NH * NS * 32) return;
    const int i  = idx & 31;
    const int s  = (idx >> 5) & (NS - 1);
    const int bh = idx >> 16;
    const int b  = bh / NH;

    const float p   = (float)pos[(size_t)b * NS + s];
    const float inv = __expf(-(float)i * (9.210340371976184f / 32.0f));
    const float ang = p * inv;
    const float cs = __cosf(ang), sn = __sinf(ang);

    const size_t base = ((size_t)bh * NS + s) * NHD + 2 * i;
    const float te = (float)t[base];
    const float to = (float)t[base + 1];
    t[base]     = (_Float16)(te * cs - to * sn);
    t[base + 1] = (_Float16)(te * sn + to * cs);
}

// ---------------------------------------------------------------------------
// Causal flash attention. Block = 256 threads = 8 waves; each wave owns a
// 16-query tile of one (b,h). Keys processed in chunks of 32.
// ---------------------------------------------------------------------------
__global__ __launch_bounds__(256) void attn_kernel(
    const _Float16* __restrict__ Qh, const _Float16* __restrict__ Kh,
    const _Float16* __restrict__ Vt, _Float16* __restrict__ Oh) {
    __shared__ __attribute__((aligned(64))) _Float16 ldsP[8][16][32];

    const int lane = threadIdx.x & 31;
    const int wv   = threadIdx.x >> 5;
    const int hf   = lane >> 4;
    const int lm   = lane & 15;

    const int bh   = blockIdx.x >> 4;        // B*H
    const int qblk = blockIdx.x & 15;
    const int q0   = qblk * 128 + wv * 16;

    const _Float16* qrow  = Qh + ((size_t)bh * NS + q0 + lm) * NHD;
    const _Float16* kbase = Kh + (size_t)bh * NS * NHD;
    const _Float16* vbase = Vt + (size_t)bh * NHD * NS;

    const v16h aq0 = load_a_frag(qrow, hf);        // hd 0..31
    const v16h aq1 = load_a_frag(qrow + 32, hf);   // hd 32..63

    float mrow[8], lrow[8];
    v8f acc[4];
#pragma unroll
    for (int r = 0; r < 8; ++r) { mrow[r] = -1e30f; lrow[r] = 0.0f; }
#pragma unroll
    for (int j = 0; j < 4; ++j) acc[j] = (v8f){};

    const int nch = (q0 + 16 + 31) >> 5;
    for (int kc = 0; kc < nch; ++kc) {
        const int k0 = kc * 32;

        // ---- scores: two 16x16 tiles over 32 keys, hd contracted in 2 WMMAs
        v8f sc[2];
#pragma unroll
        for (int t = 0; t < 2; ++t) {
            const _Float16* krow = kbase + (size_t)(k0 + t * 16 + lm) * NHD;
            v8f s = {};
            s = wmma_f16(aq0, *reinterpret_cast<const v16h*>(krow + hf * 16), s);
            s = wmma_f16(aq1, *reinterpret_cast<const v16h*>(krow + 32 + hf * 16), s);
            sc[t] = s * 0.125f;                   // 1/sqrt(64)
        }

        // ---- causal mask + row max
        float mc[8];
#pragma unroll
        for (int r = 0; r < 8; ++r) {
            const int q = q0 + r + hf * 8;
            if (k0 + lm > q)      sc[0][r] = -1e30f;
            if (k0 + 16 + lm > q) sc[1][r] = -1e30f;
            mc[r] = fmaxf(sc[0][r], sc[1][r]);
        }
#pragma unroll
        for (int r = 0; r < 8; ++r)
            for (int off = 1; off < 16; off <<= 1)
                mc[r] = fmaxf(mc[r], __shfl_xor(mc[r], off, 32));

        float al[8], rs[8];
#pragma unroll
        for (int r = 0; r < 8; ++r) {
            const float mn = fmaxf(mrow[r], mc[r]);
            al[r] = __expf(mrow[r] - mn);
            mrow[r] = mn;
            sc[0][r] = __expf(sc[0][r] - mn);
            sc[1][r] = __expf(sc[1][r] - mn);
            rs[r] = sc[0][r] + sc[1][r];
        }
#pragma unroll
        for (int r = 0; r < 8; ++r)
            for (int off = 1; off < 16; off <<= 1)
                rs[r] += __shfl_xor(rs[r], off, 32);
#pragma unroll
        for (int r = 0; r < 8; ++r) lrow[r] = lrow[r] * al[r] + rs[r];

        // ---- transpose P from C-layout to A-layout via LDS
#pragma unroll
        for (int r = 0; r < 8; ++r) {
            ldsP[wv][r + hf * 8][lm]      = (_Float16)sc[0][r];
            ldsP[wv][r + hf * 8][16 + lm] = (_Float16)sc[1][r];
        }
        asm volatile("s_wait_dscnt 0" ::: "memory");
        const v16h ap = load_a_frag(&ldsP[wv][lm][0], hf);

        // ---- PV: rescale accumulators, 4 WMMAs over the 64 hd columns
#pragma unroll
        for (int j = 0; j < 4; ++j) {
#pragma unroll
            for (int r = 0; r < 8; ++r) acc[j][r] *= al[r];
            const _Float16* vrow =
                vbase + (size_t)(j * 16 + lm) * NS + k0 + hf * 16;
            acc[j] = wmma_f16(ap, *reinterpret_cast<const v16h*>(vrow), acc[j]);
        }
    }

    // ---- normalize and write out in [b][s][D] fp16
    const int b = bh / NH;
    const int h = bh - b * NH;
#pragma unroll
    for (int j = 0; j < 4; ++j) {
#pragma unroll
        for (int r = 0; r < 8; ++r) {
            const int s = q0 + r + hf * 8;
            Oh[((size_t)b * NS + s) * ND + h * NHD + j * 16 + lm] =
                (_Float16)(acc[j][r] / lrow[r]);
        }
    }
}

// ---------------------------------------------------------------------------
// Output projection: out[m][o] = sum_d Oh[m][d] * W_O[o][d], fp32 out.
// Each wave computes a 16x64 strip; fully-unrolled K loop for pipelining.
// ---------------------------------------------------------------------------
__global__ __launch_bounds__(256) void out_gemm_kernel(
    const _Float16* __restrict__ Oh, const _Float16* __restrict__ woh,
    float* __restrict__ out) {
    const int lane = threadIdx.x & 31;
    const int wv   = threadIdx.x >> 5;
    const int hf   = lane >> 4;
    const int lm   = lane & 15;

    int tile = blockIdx.x * 8 + wv;          // 512 * 12 strips
    const int nt4 = tile % 12;
    const int mt  = tile / 12;

    const _Float16* arow  = Oh + (size_t)(mt * 16 + lm) * ND;
    const _Float16* bbase = woh + (size_t)(nt4 * 64 + lm) * ND;

    v8f acc[4];
#pragma unroll
    for (int n = 0; n < 4; ++n) acc[n] = (v8f){};

#pragma unroll
    for (int c = 0; c < KCH; ++c) {
        const v16h a = load_a_frag(arow + c * 32, hf);
#pragma unroll
        for (int n = 0; n < 4; ++n) {
            const v16h b = *reinterpret_cast<const v16h*>(
                bbase + (size_t)(n * 16) * ND + c * 32 + hf * 16);
            acc[n] = wmma_f16(a, b, acc[n]);
        }
    }

#pragma unroll
    for (int n = 0; n < 4; ++n)
#pragma unroll
        for (int r = 0; r < 8; ++r)
            out[(size_t)(mt * 16 + r + hf * 8) * ND + nt4 * 64 + n * 16 + lm] =
                acc[n][r];
}

// ---------------------------------------------------------------------------
// Host-side launcher
// ---------------------------------------------------------------------------
extern "C" void kernel_launch(void* const* d_in, const int* in_sizes, int n_in,
                              void* d_out, int out_size, void* d_ws,
                              size_t ws_size, hipStream_t stream) {
    const float*     x    = (const float*)d_in[0];
    const float*     wqkv = (const float*)d_in[1];
    const float*     wo   = (const float*)d_in[2];
    const long long* tpos = (const long long*)d_in[3];
    float*           out  = (float*)d_out;

    constexpr size_t nX    = (size_t)NB * NS * ND;   // 6291456
    constexpr size_t nWqkv = (size_t)3 * ND * ND;    // 1769472
    constexpr size_t nWo   = (size_t)ND * ND;        // 589824
    constexpr size_t nHead = (size_t)NB * NH * NS * NHD;  // == nX

    auto align256 = [](size_t v) { return (v + 255) & ~(size_t)255; };
    uint8_t* ws = (uint8_t*)d_ws;
    size_t off = 0;
    _Float16* xh  = (_Float16*)(ws + off); off = align256(off + nX * 2);
    _Float16* wqh = (_Float16*)(ws + off); off = align256(off + nWqkv * 2);
    _Float16* woh = (_Float16*)(ws + off); off = align256(off + nWo * 2);
    _Float16* Qh  = (_Float16*)(ws + off); off = align256(off + nHead * 2);
    _Float16* Kh  = (_Float16*)(ws + off); off = align256(off + nHead * 2);
    _Float16* Vt  = (_Float16*)(ws + off); off = align256(off + nHead * 2);
    _Float16* Oh  = (_Float16*)(ws + off); off = align256(off + nX * 2);
    (void)ws_size; (void)in_sizes; (void)n_in; (void)out_size;

    f32_to_f16_kernel<<<(int)((nX + 255) / 256), 256, 0, stream>>>(x, xh, (int)nX);
    f32_to_f16_kernel<<<(int)((nWqkv + 255) / 256), 256, 0, stream>>>(wqkv, wqh, (int)nWqkv);
    f32_to_f16_kernel<<<(int)((nWo + 255) / 256), 256, 0, stream>>>(wo, woh, (int)nWo);

    // 3 * (8192/16) * (768/64) = 18432 strips, 8 waves per block
    qkv_gemm_kernel<<<2304, 256, 0, stream>>>(xh, wqh, Qh, Kh, Vt);

    const int nRope = NB * NH * NS * 32;  // 3145728
    rope_kernel<<<nRope / 256, 256, 0, stream>>>(Qh, tpos);
    rope_kernel<<<nRope / 256, 256, 0, stream>>>(Kh, tpos);

    // B*H * (S/128) = 768 blocks
    attn_kernel<<<NB * NH * (NS / 128), 256, 0, stream>>>(Qh, Kh, Vt, Oh);

    // (8192/16) * (768/64) = 6144 strips, 8 per block
    out_gemm_kernel<<<768, 256, 0, stream>>>(Oh, woh, out);
}
